// SpMiddleFHD_197568495909
// MI455X (gfx1250) — compile-verified, matrix operand order checked by reading, not run on hardware
//
#include <hip/hip_runtime.h>

typedef __attribute__((ext_vector_type(16))) __bf16 v16bf;
typedef __attribute__((ext_vector_type(8)))  __bf16 v8bf;
typedef __attribute__((ext_vector_type(8)))  float  v8f;

#define BN_EPS 0.001f

// ---------------------------------------------------------------------------
// CDNA5 16-bit A/B fragment K-map (wave32), per ISA 7.12.2:
// lanes 0-15 : j 0..7 -> K=0..7   ; j 8..15 -> K=16..23
// lanes16-31 : j 0..7 -> K=8..15  ; j 8..15 -> K=24..31
// ---------------------------------------------------------------------------
__device__ __forceinline__ int kmapK(int lane, int j) {
    int r = j >> 1, h = j & 1;
    int hi = (lane >> 4) & 1;
    int base = (r < 4) ? ((hi ? 8 : 0) + 2 * r) : ((hi ? 24 : 16) + 2 * (r - 4));
    return base + h;
}

// ---------------------------------------------------------------------------
// Scatter voxel features into dense bf16 grid + mask
// ---------------------------------------------------------------------------
__global__ void scatter_kernel(const float* __restrict__ feats, const int* __restrict__ coors,
                               __bf16* __restrict__ x, float* __restrict__ mask,
                               int N, int D, int H, int W, int C) {
    int n = blockIdx.x * blockDim.x + threadIdx.x;
    if (n >= N) return;
    int b = coors[n * 4 + 0], z = coors[n * 4 + 1];
    int y = coors[n * 4 + 2], xw = coors[n * 4 + 3];
    long base = (((long)b * D + z) * H + y) * (long)W + xw;
    for (int c = 0; c < C; ++c) x[base * C + c] = (__bf16)feats[n * C + c];
    mask[base] = 1.0f;
}

// ---------------------------------------------------------------------------
// Downsample mask: out = any(in window) > 0
// ---------------------------------------------------------------------------
__global__ void mask_down_kernel(const float* __restrict__ min_, float* __restrict__ mout,
                                 int B, int Di, int Hi, int Wi, int Do, int Ho, int Wo,
                                 int kd, int kh, int kw, int sd, int sh, int sw,
                                 int pd, int ph, int pw) {
    long i = blockIdx.x * (long)blockDim.x + threadIdx.x;
    long M = (long)B * Do * Ho * Wo;
    if (i >= M) return;
    long r = i;
    int w_ = (int)(r % Wo); r /= Wo;
    int h_ = (int)(r % Ho); r /= Ho;
    int d_ = (int)(r % Do); r /= Do;
    int b  = (int)r;
    float v = 0.f;
    for (int kz = 0; kz < kd; ++kz) {
        int iz = d_ * sd - pd + kz; if ((unsigned)iz >= (unsigned)Di) continue;
        for (int ky = 0; ky < kh; ++ky) {
            int iy = h_ * sh - ph + ky; if ((unsigned)iy >= (unsigned)Hi) continue;
            for (int kx = 0; kx < kw; ++kx) {
                int ix = w_ * sw - pw + kx; if ((unsigned)ix >= (unsigned)Wi) continue;
                v += min_[(((long)b * Di + iz) * Hi + iy) * (long)Wi + ix];
            }
        }
    }
    mout[i] = v > 0.f ? 1.f : 0.f;
}

// ---------------------------------------------------------------------------
// One contiguous 8-half (b128) A run: GEMM-K index g -> (tap, ci), bounds-
// checked against the padded input volume.
// ---------------------------------------------------------------------------
template <int CI>
__device__ __forceinline__ v8bf loadRunA(const __bf16* __restrict__ xin,
                                         int g, int totalK, int kw, int kh,
                                         int iz0, int iy0, int ix0,
                                         int Di, int Hi, int Wi,
                                         long vb, bool mvalid) {
    v8bf z = {};
    if (!mvalid || g >= totalK) return z;
    int tap = g / CI;          // compile-time shift (CI = 16/32/64)
    int ci  = g & (CI - 1);
    int kx = tap % kw;
    int tk = tap / kw;
    int ky = tk % kh;
    int kz = tk / kh;
    int iz = iz0 + kz, iy = iy0 + ky, ix = ix0 + kx;
    if ((unsigned)iz >= (unsigned)Di || (unsigned)iy >= (unsigned)Hi ||
        (unsigned)ix >= (unsigned)Wi)
        return z;
    long base = (((vb * Di + iz) * Hi + iy) * (long)Wi + ix) * CI + ci;
    return *(const v8bf*)(xin + base);
}

// ---------------------------------------------------------------------------
// Implicit-GEMM 3D conv, bf16 WMMA (fp32 acc), templated on (Ci, Co).
// GEMM: M = output voxels, N = Co, K = nk*Ci flattened (tap-major, ci inner)
// so weights are addressed as w[g*Co + co] directly. 32-wide K chunks are
// always dense (Ci=16 packs two taps per chunk).
// Block = 256 threads = 8 waves; wave -> 16 voxels x ALL Co channels:
// one A fragment feeds NT = Co/16 back-to-back WMMAs (register N-blocking).
// ---------------------------------------------------------------------------
template <int CI, int CO>
__global__ void conv_wmma_kernel(const __bf16* __restrict__ xin, const float* __restrict__ w,
                                 const float* __restrict__ maskOut, __bf16* __restrict__ yout,
                                 int B, int Di, int Hi, int Wi,
                                 int Do, int Ho, int Wo,
                                 int kd, int kh, int kw,
                                 int sd, int sh, int sw,
                                 int pd, int ph, int pw) {
    constexpr int NT = CO / 16;                 // N tiles per wave
    constexpr int E  = CO * 32 / 256;           // halfs staged per thread
    __shared__ __align__(32) __bf16 ldsB[NT * 512];  // NT x (32 lane-slots x 16 halfs)

    const int t    = threadIdx.x;
    const int lane = t & 31;
    const int wave = t >> 5;
    const int col  = lane & 15;
    const int hi   = lane >> 4;

    const long M  = (long)B * Do * Ho * Wo;
    const long m0 = ((long)blockIdx.x * 8 + wave) * 16;

    // decompose this lane's A-row voxel
    long mrow = m0 + col;
    int vb = 0, vd = 0, vh = 0, vw = 0;
    bool mvalid = mrow < M;
    if (mvalid) {
        long r = mrow;
        vw = (int)(r % Wo); r /= Wo;
        vh = (int)(r % Ho); r /= Ho;
        vd = (int)(r % Do); r /= Do;
        vb = (int)r;
    }
    const int iz0 = vd * sd - pd, iy0 = vh * sh - ph, ix0 = vw * sw - pw;

    v8f acc[NT];
#pragma unroll
    for (int n = 0; n < NT; ++n) acc[n] = (v8f){};

    const int nk     = kd * kh * kw;
    const int totalK = nk * CI;
    const int NC     = (totalK + 31) >> 5;
    const int off0   = hi ? 8 : 0;              // lane's A run start in chunk

    for (int c = 0; c < NC; ++c) {
        const int gbase = c << 5;

        // --- stage B tiles for all NT column blocks (CO*32 halfs total),
        //     fragment layout; weight address is simply g*CO + co.
        {
#pragma unroll
            for (int q = 0; q < E; ++q) {
                int f   = t * E + q;            // flat half index in [0, CO*32)
                int nt  = f >> 9;
                int rem = f & 511;
                int bl  = rem >> 4;
                int j   = rem & 15;
                int g   = gbase + kmapK(bl, j);
                int co  = nt * 16 + (bl & 15);
                float wv = (g < totalK) ? w[(long)g * CO + co] : 0.0f;
                ldsB[nt * 512 + bl * 16 + j] = (__bf16)wv;
            }
        }
        __syncthreads();

        // --- A fragment: two aligned b128 runs per lane
        v8bf lo  = loadRunA<CI>(xin, gbase + off0,      totalK, kw, kh,
                                iz0, iy0, ix0, Di, Hi, Wi, (long)vb, mvalid);
        v8bf hiv = loadRunA<CI>(xin, gbase + off0 + 16, totalK, kw, kh,
                                iz0, iy0, ix0, Di, Hi, Wi, (long)vb, mvalid);
        v16bf afrag;
#pragma unroll
        for (int j = 0; j < 8; ++j) {
            afrag[j]     = lo[j];
            afrag[j + 8] = hiv[j];
        }

        // --- NT WMMAs reuse the single A fragment
#pragma unroll
        for (int n = 0; n < NT; ++n) {
            v16bf bfrag = *((const v16bf*)&ldsB[n * 512 + lane * 16]);
            acc[n] = __builtin_amdgcn_wmma_f32_16x16x32_bf16(
                false, afrag, false, bfrag, (short)0, acc[n], false, false);
        }
        __syncthreads();
    }

    // store: lane holds voxel m0 + v + hi*8, channel n*16+col, for v=0..7
#pragma unroll
    for (int v = 0; v < 8; ++v) {
        long mo = m0 + v + hi * 8;
        if (mo < M) {
            float mval = maskOut[mo];
#pragma unroll
            for (int n = 0; n < NT; ++n)
                yout[mo * CO + n * 16 + col] = (__bf16)(acc[n][v] * mval);
        }
    }
}

// ---------------------------------------------------------------------------
// Masked BN stats: stats[0..63]=sum, stats[64..127]=sumsq, stats[128]=count
// Co is a power of two: lco = log2(Co).
// ---------------------------------------------------------------------------
__global__ void bn_reduce_kernel(const __bf16* __restrict__ y, const float* __restrict__ mask,
                                 float* __restrict__ stats, long nvox, int Co, int lco) {
    __shared__ float ls[256], ls2[256], lc[256];
    int t = threadIdx.x;
    int c = t & (Co - 1);
    float s = 0.f, s2 = 0.f, cnt = 0.f;
    long total = nvox << lco;
    for (long i = (long)blockIdx.x * 256 + t; i < total; i += 256L * gridDim.x) {
        long v  = i >> lco;
        float m = mask[v];
        float xv = (float)y[i];
        s  += xv * m;
        s2 += xv * xv * m;
        if (c == 0) cnt += m;
    }
    ls[t] = s; ls2[t] = s2; lc[t] = cnt;
    __syncthreads();
    for (int off = 128; off >= Co; off >>= 1) {
        if (t < off) { ls[t] += ls[t + off]; ls2[t] += ls2[t + off]; lc[t] += lc[t + off]; }
        __syncthreads();
    }
    if (t < Co) {
        atomicAdd(&stats[c],      ls[t]);
        atomicAdd(&stats[64 + c], ls2[t]);
    }
    if (t == 0) atomicAdd(&stats[128], lc[0]);
}

__global__ void bn_apply_kernel(__bf16* __restrict__ y, const float* __restrict__ mask,
                                const float* __restrict__ stats,
                                const float* __restrict__ gamma, const float* __restrict__ beta,
                                long nvox, int Co, int lco) {
    long i = blockIdx.x * (long)blockDim.x + threadIdx.x;
    long total = nvox << lco;
    if (i >= total) return;
    int  c = (int)(i & (Co - 1));
    long v = i >> lco;
    float cnt = stats[128]; cnt = cnt > 1.f ? cnt : 1.f;
    float mean = stats[c] / cnt;
    float var  = stats[64 + c] / cnt - mean * mean;
    float inv  = rsqrtf(var + BN_EPS);
    float xv   = (float)y[i];
    float val  = (xv - mean) * inv * gamma[c] + beta[c];
    val = val > 0.f ? val : 0.f;
    y[i] = (__bf16)(val * mask[v]);
}

// ---------------------------------------------------------------------------
// Final: (B,D,H,W,C) bf16 -> (B, C*D, H, W) fp32
// ---------------------------------------------------------------------------
__global__ void final_out_kernel(const __bf16* __restrict__ x, float* __restrict__ out,
                                 int B, int Dn, int Hn, int Wn, int Cn) {
    long i = blockIdx.x * (long)blockDim.x + threadIdx.x;
    long total = (long)B * Cn * Dn * Hn * Wn;
    if (i >= total) return;
    long r = i;
    int w_ = (int)(r % Wn); r /= Wn;
    int h_ = (int)(r % Hn); r /= Hn;
    int d_ = (int)(r % Dn); r /= Dn;
    int c  = (int)(r % Cn); r /= Cn;
    int b  = (int)r;
    out[i] = (float)x[((((long)b * Dn + d_) * Hn + h_) * (long)Wn + w_) * Cn + c];
}

// ---------------------------------------------------------------------------
// Host orchestration
// ---------------------------------------------------------------------------
struct LayerDef { int down, co, kd, kh, kw, sd, sh, sw, pd, ph, pw; };

static void launch_conv(int Ci, dim3 grid, hipStream_t stream,
                        const __bf16* cur, const float* w, const float* mcur, __bf16* nxt,
                        int B, int Di, int Hi, int Wi,
                        int Do, int Ho, int Wo, const LayerDef& d) {
#define CONV_ARGS cur, w, mcur, nxt, B, Di, Hi, Wi, Do, Ho, Wo, \
                  d.kd, d.kh, d.kw, d.sd, d.sh, d.sw, d.pd, d.ph, d.pw
    if (Ci == 16 && d.co == 16)
        conv_wmma_kernel<16, 16><<<grid, 256, 0, stream>>>(CONV_ARGS);
    else if (Ci == 16 && d.co == 32)
        conv_wmma_kernel<16, 32><<<grid, 256, 0, stream>>>(CONV_ARGS);
    else if (Ci == 32 && d.co == 32)
        conv_wmma_kernel<32, 32><<<grid, 256, 0, stream>>>(CONV_ARGS);
    else if (Ci == 32 && d.co == 64)
        conv_wmma_kernel<32, 64><<<grid, 256, 0, stream>>>(CONV_ARGS);
    else
        conv_wmma_kernel<64, 64><<<grid, 256, 0, stream>>>(CONV_ARGS);
#undef CONV_ARGS
}

extern "C" void kernel_launch(void* const* d_in, const int* in_sizes, int n_in,
                              void* d_out, int out_size, void* d_ws, size_t ws_size,
                              hipStream_t stream) {
    const float* feats = (const float*)d_in[0];
    const int*   coors = (const int*)d_in[1];
    const int    B = 2;                       // fixed by harness setup
    const int    N = in_sizes[0] / 16;

    const int NL = 14;
    const float *wts[NL], *gms[NL], *bts[NL];
    for (int i = 0; i < NL; ++i) {
        wts[i] = (const float*)d_in[3 + i];
        gms[i] = (const float*)d_in[3 + NL + i];
        bts[i] = (const float*)d_in[3 + 2 * NL + i];
    }

    static const LayerDef L[NL] = {
        {0, 16, 3,3,3, 1,1,1, 1,1,1},
        {0, 16, 3,3,3, 1,1,1, 1,1,1},
        {1, 32, 3,3,3, 2,2,2, 1,1,1},
        {0, 32, 3,3,3, 1,1,1, 1,1,1},
        {0, 32, 3,3,3, 1,1,1, 1,1,1},
        {1, 64, 3,3,3, 2,2,2, 1,1,1},
        {0, 64, 3,3,3, 1,1,1, 1,1,1},
        {0, 64, 3,3,3, 1,1,1, 1,1,1},
        {0, 64, 3,3,3, 1,1,1, 1,1,1},
        {1, 64, 3,3,3, 2,2,2, 0,1,1},
        {0, 64, 3,3,3, 1,1,1, 1,1,1},
        {0, 64, 3,3,3, 1,1,1, 1,1,1},
        {0, 64, 3,3,3, 1,1,1, 1,1,1},
        {1, 64, 3,1,1, 2,1,1, 0,0,0},
    };

    // workspace layout
    const long maxAct = (long)B * 41 * 256 * 224 * 16;   // level-0 elems
    const long maxVox = (long)B * 41 * 256 * 224;
    char* ws = (char*)d_ws;
    __bf16* actA = (__bf16*)ws;                ws += ((maxAct * 2 + 255) / 256) * 256;
    __bf16* actB = (__bf16*)ws;                ws += ((maxAct * 2 + 255) / 256) * 256;
    float*  maskA = (float*)ws;                ws += ((maxVox * 4 + 255) / 256) * 256;
    float*  maskB = (float*)ws;                ws += ((maxVox * 4 + 255) / 256) * 256;
    float*  stats = (float*)ws;                ws += 256 * 4;

    hipMemsetAsync(actA,  0, (size_t)maxAct * 2, stream);
    hipMemsetAsync(maskA, 0, (size_t)maxVox * 4, stream);

    scatter_kernel<<<(N + 255) / 256, 256, 0, stream>>>(
        feats, coors, actA, maskA, N, 41, 256, 224, 16);

    int Di = 41, Hi = 256, Wi = 224, Ci = 16;
    __bf16 *cur = actA, *nxt = actB;
    float  *mcur = maskA, *mnxt = maskB;

    for (int l = 0; l < NL; ++l) {
        const LayerDef& d = L[l];
        int Do = Di, Ho = Hi, Wo = Wi;
        if (d.down) {
            Do = (Di + 2 * d.pd - d.kd) / d.sd + 1;
            Ho = (Hi + 2 * d.ph - d.kh) / d.sh + 1;
            Wo = (Wi + 2 * d.pw - d.kw) / d.sw + 1;
            long Mo = (long)B * Do * Ho * Wo;
            mask_down_kernel<<<(unsigned)((Mo + 255) / 256), 256, 0, stream>>>(
                mcur, mnxt, B, Di, Hi, Wi, Do, Ho, Wo,
                d.kd, d.kh, d.kw, d.sd, d.sh, d.sw, d.pd, d.ph, d.pw);
            float* tmp = mcur; mcur = mnxt; mnxt = tmp;
        }
        long M = (long)B * Do * Ho * Wo;
        dim3 grid((unsigned)((M + 127) / 128));
        launch_conv(Ci, grid, stream, cur, wts[l], mcur, nxt,
                    B, Di, Hi, Wi, Do, Ho, Wo, d);

        hipMemsetAsync(stats, 0, 192 * 4, stream);
        int lco = (d.co == 16) ? 4 : (d.co == 32) ? 5 : 6;
        long total = M * d.co;
        long rb = (total + 255) / 256; if (rb > 2048) rb = 2048; if (rb < 1) rb = 1;
        bn_reduce_kernel<<<(unsigned)rb, 256, 0, stream>>>(nxt, mcur, stats, M, d.co, lco);
        bn_apply_kernel<<<(unsigned)((total + 255) / 256), 256, 0, stream>>>(
            nxt, mcur, stats, gms[l], bts[l], M, d.co, lco);

        __bf16* t = cur; cur = nxt; nxt = t;
        Di = Do; Hi = Ho; Wi = Wo; Ci = d.co;
    }

    long totalOut = (long)B * Ci * Di * Hi * Wi;
    final_out_kernel<<<(unsigned)((totalOut + 255) / 256), 256, 0, stream>>>(
        cur, (float*)d_out, B, Di, Hi, Wi, Ci);
}
